// customClass_26096221291172
// MI455X (gfx1250) — compile-verified, matrix operand order checked by reading, not run on hardware
//
#include <hip/hip_runtime.h>

// ---------- types ----------
typedef __attribute__((ext_vector_type(16))) __bf16          v16bf;
typedef __attribute__((ext_vector_type(8)))  float           v8f;
typedef __attribute__((ext_vector_type(8)))  unsigned short  ush8;

union BF16x16 { v16bf v; ush8 h[2]; };

#define DK 256        // feature dim
#define NQ 8192       // query rows
#define NC 8192       // context rows
#define NCHUNK 4      // context chunks (occupancy)
#define STAGE_ROWS 32 // context rows per LDS stage (double buffered)

// float -> bf16, round-to-nearest-even
__device__ __forceinline__ unsigned short f2bf(float x) {
    unsigned u = __float_as_uint(x);
    u += 0x7FFFu + ((u >> 16) & 1u);
    return (unsigned short)(u >> 16);
}

__device__ __forceinline__ v8f wmma_bf16(v16bf a, v16bf b, v8f c) {
    return __builtin_amdgcn_wmma_f32_16x16x32_bf16(
        /*neg_a=*/false, a, /*neg_b=*/false, b,
        /*c_mod=*/(short)0, c, /*reuse_a=*/false, /*reuse_b=*/false);
}

// Async 16B global -> LDS copy, tracked by ASYNCcnt (ISA 08 §4).
// VDST vgpr carries the per-lane LDS byte address (low 32 bits of flat ptr).
__device__ __forceinline__ void async_b128(const void* g, void* l) {
    asm volatile("global_load_async_to_lds_b128 %0, %1, off"
                 :: "v"((unsigned)(unsigned long long)l), "v"(g)
                 : "memory");
}

// Issue one stage's copies: 2 arrays x 16 KB = 1024 x 16B chunks, 4 per thread
__device__ __forceinline__ void stage_copy(const unsigned short* __restrict__ gr,
                                           const unsigned short* __restrict__ gn,
                                           unsigned short* lr, unsigned short* ln,
                                           int tid) {
#pragma unroll
    for (int i = 0; i < 4; ++i) {
        int idx = (tid + i * 256) * 8;   // ushort index of 16B chunk
        async_b128(gr + idx, lr + idx);
        async_b128(gn + idx, ln + idx);
    }
}

// Load a 16x32 bf16 WMMA operand fragment from a row-major [rows][256] bf16
// array.  Per ISA 7.12.2: lane L (0..15) holds row L, K = {0..7, 16..23};
// lane L+16 holds row L, K = {8..15, 24..31}.  B operand mirrors with
// lane = column = context row, so identical loading gives A * B^T.
__device__ __forceinline__ v16bf load_frag(const unsigned short* __restrict__ base,
                                           int row0, int k0, int lane) {
    int r = row0 + (lane & 15);
    int c = k0 + ((lane >> 4) & 1) * 8;
    const unsigned short* p = base + (size_t)r * DK + c;
    BF16x16 u;
    u.h[0] = *(const ush8*)(p);        // K +0..7
    u.h[1] = *(const ush8*)(p + 16);   // K +16..23
    return u.v;
}

// ---------- prep: row L2 norms + bf16 conversion (raw and normalized) -------
__global__ __launch_bounds__(256) void prep_rows(const float* __restrict__ src,
                                                 unsigned short* __restrict__ raw,
                                                 unsigned short* __restrict__ nrm,
                                                 int R) {
    int wave = threadIdx.x >> 5, lane = threadIdx.x & 31;
    int row = blockIdx.x * 8 + wave;
    if (row >= R) return;
    const float4* p = (const float4*)(src + (size_t)row * DK + lane * 8);
    float4 a = p[0], b = p[1];
    float s = a.x*a.x + a.y*a.y + a.z*a.z + a.w*a.w
            + b.x*b.x + b.y*b.y + b.z*b.z + b.w*b.w;
#pragma unroll
    for (int m = 16; m >= 1; m >>= 1) s += __shfl_xor(s, m, 32);
    float inv = rsqrtf(s);
    float va[8] = {a.x, a.y, a.z, a.w, b.x, b.y, b.z, b.w};
    ush8 r8, n8;
#pragma unroll
    for (int i = 0; i < 8; ++i) { r8[i] = f2bf(va[i]); n8[i] = f2bf(va[i] * inv); }
    size_t base = (size_t)row * DK + lane * 8;
    *(ush8*)(raw + base) = r8;
    *(ush8*)(nrm + base) = n8;
}

// ---------- prep: W fp32 -> bf16 ----------
__global__ __launch_bounds__(256) void wconv(const float* __restrict__ w,
                                             unsigned short* __restrict__ wb) {
    int i = blockIdx.x * 256 + threadIdx.x;
    wb[i] = f2bf(w[i]);
}

// ---------- q_lin = query @ W^T via WMMA ----------
__global__ __launch_bounds__(256) void qlin_wmma(const unsigned short* __restrict__ qr,
                                                 const unsigned short* __restrict__ wb,
                                                 unsigned short* __restrict__ ql) {
    int wave = threadIdx.x >> 5, lane = threadIdx.x & 31;
    int row0 = (blockIdx.x * 8 + wave) * 16;
    v16bf a[8];
#pragma unroll
    for (int k = 0; k < 8; ++k) a[k] = load_frag(qr, row0, k * 32, lane);
    for (int ot = 0; ot < 16; ++ot) {
        v8f acc = {};
#pragma unroll
        for (int k = 0; k < 8; ++k) {
            v16bf b = load_frag(wb, ot * 16, k * 32, lane);
            acc = wmma_bf16(a[k], b, acc);
        }
        // C/D layout: lanes 0-15 VGPR v -> (M=v, N=lane); lanes 16-31 -> (M=v+8)
        int col = ot * 16 + (lane & 15);
        int rbase = row0 + ((lane >> 4) << 3);
#pragma unroll
        for (int v = 0; v < 8; ++v)
            ql[(size_t)(rbase + v) * DK + col] = f2bf(acc[v]);
    }
}

// ---------- fused dual-GEMM + online softmax, async double-buffered ----------
__global__ __launch_bounds__(256) void attn_flash(
        const unsigned short* __restrict__ qn, const unsigned short* __restrict__ ql,
        const unsigned short* __restrict__ cn, const unsigned short* __restrict__ cr,
        float* __restrict__ pm, float* __restrict__ pl, float* __restrict__ pa) {
    __shared__ unsigned short lbr[2][STAGE_ROWS * DK];  // raw ctx, 2 x 16 KB
    __shared__ unsigned short lbn[2][STAGE_ROWS * DK];  // norm ctx, 2 x 16 KB
    int wave = threadIdx.x >> 5, lane = threadIdx.x & 31;
    int tid = threadIdx.x;
    int row0 = (blockIdx.x * 8 + wave) * 16;
    int chunk = blockIdx.y;
    size_t cstart = (size_t)chunk * (NC / NCHUNK);
    const int nstage = (NC / NCHUNK) / STAGE_ROWS;

    // preload this wave's query fragments (all K) into registers
    v16bf aq[8], an[8];
#pragma unroll
    for (int k = 0; k < 8; ++k) {
        aq[k] = load_frag(ql, row0, k * 32, lane);
        an[k] = load_frag(qn, row0, k * 32, lane);
    }
    float m[8], l[8], acc[8];
#pragma unroll
    for (int v = 0; v < 8; ++v) { m[v] = -1e30f; l[v] = 0.f; acc[v] = 0.f; }

    // prologue: kick off stage 0 copy
    stage_copy(cr + cstart * DK, cn + cstart * DK, lbr[0], lbn[0], tid);

    for (int stage = 0; stage < nstage; ++stage) {
        int b = stage & 1;
        if (stage + 1 < nstage) {
            // start next stage into the other buffer (freed by the barrier
            // at the end of the previous iteration)
            size_t nb = (cstart + (size_t)(stage + 1) * STAGE_ROWS) * DK;
            stage_copy(cr + nb, cn + nb, lbr[b ^ 1], lbn[b ^ 1], tid);
            // oldest 8 async ops (current stage) must have landed
            asm volatile("s_wait_asynccnt 0x8" ::: "memory");
        } else {
            asm volatile("s_wait_asynccnt 0x0" ::: "memory");
        }
        __syncthreads();   // everyone's copies for buffer b are complete
#pragma unroll
        for (int ct = 0; ct < STAGE_ROWS / 16; ++ct) {
            v8f S = {}, C = {};
#pragma unroll
            for (int k = 0; k < 8; ++k) {
                v16bf br = load_frag(lbr[b], ct * 16, k * 32, lane);
                S = wmma_bf16(aq[k], br, S);               // scores tile
                v16bf bn = load_frag(lbn[b], ct * 16, k * 32, lane);
                C = wmma_bf16(an[k], bn, C);               // cosine tile
            }
            // online softmax update; row state lives across the 16 lanes
            // holding that row's columns (xor masks 1..8 stay in-half)
#pragma unroll
            for (int v = 0; v < 8; ++v) {
                float s = S[v];
                float t = fmaxf(s, __shfl_xor(s, 1, 32));
                t = fmaxf(t, __shfl_xor(t, 2, 32));
                t = fmaxf(t, __shfl_xor(t, 4, 32));
                t = fmaxf(t, __shfl_xor(t, 8, 32));
                float mn = fmaxf(m[v], t);
                float f  = __expf(m[v] - mn);
                float p  = __expf(s - mn);
                l[v]   = fmaf(l[v], f, p);
                acc[v] = fmaf(acc[v], f, p * C[v]);
                m[v]   = mn;
            }
        }
        __syncthreads();   // buffer b free for reuse next iteration
    }
    // reduce lane-partial l/acc across the 16 columns, emit chunk partials
#pragma unroll
    for (int v = 0; v < 8; ++v) {
        float lv = l[v], av = acc[v];
        lv += __shfl_xor(lv, 1, 32); av += __shfl_xor(av, 1, 32);
        lv += __shfl_xor(lv, 2, 32); av += __shfl_xor(av, 2, 32);
        lv += __shfl_xor(lv, 4, 32); av += __shfl_xor(av, 4, 32);
        lv += __shfl_xor(lv, 8, 32); av += __shfl_xor(av, 8, 32);
        if ((lane & 15) == 0) {
            int row = row0 + v + ((lane >> 4) << 3);
            size_t p = (size_t)row * NCHUNK + chunk;
            pm[p] = m[v]; pl[p] = lv; pa[p] = av;
        }
    }
}

// ---------- deterministic log-sum-exp merge over chunks ----------
__global__ __launch_bounds__(256) void merge_chunks(const float* __restrict__ pm,
                                                    const float* __restrict__ pl,
                                                    const float* __restrict__ pa,
                                                    float* __restrict__ yrow) {
    int i = blockIdx.x * 256 + threadIdx.x;
    float M = -1e30f;
#pragma unroll
    for (int c = 0; c < NCHUNK; ++c) M = fmaxf(M, pm[(size_t)i * NCHUNK + c]);
    float L = 0.f, A = 0.f;
#pragma unroll
    for (int c = 0; c < NCHUNK; ++c) {
        float f = __expf(pm[(size_t)i * NCHUNK + c] - M);
        L = fmaf(pl[(size_t)i * NCHUNK + c], f, L);
        A = fmaf(pa[(size_t)i * NCHUNK + c], f, A);
    }
    yrow[i] = A / L;
}

// ---------- final fixed-order reduction + mean + ReLU ----------
__global__ __launch_bounds__(256) void final_reduce(const float* __restrict__ yrow,
                                                    float* __restrict__ out) {
    __shared__ float sm[256];
    int t = threadIdx.x;
    float s = 0.f;
    for (int i = t; i < NQ; i += 256) s += yrow[i];
    sm[t] = s;
    __syncthreads();
    for (int off = 128; off > 0; off >>= 1) {
        if (t < off) sm[t] += sm[t + off];
        __syncthreads();
    }
    if (t == 0) out[0] = fmaxf(sm[0] / (float)NQ, 0.0f);
}

// ---------- launch ----------
extern "C" void kernel_launch(void* const* d_in, const int* in_sizes, int n_in,
                              void* d_out, int out_size, void* d_ws, size_t ws_size,
                              hipStream_t stream) {
    const float* query   = (const float*)d_in[0];   // [8192, 256]
    const float* context = (const float*)d_in[1];   // [8192, 256]
    const float* W       = (const float*)d_in[2];   // [256, 256]
    float* out = (float*)d_out;

    unsigned short* qr = (unsigned short*)d_ws;          // raw query bf16
    unsigned short* qn = qr + (size_t)NQ * DK;           // normalized query bf16
    unsigned short* ql = qn + (size_t)NQ * DK;           // q_lin bf16
    unsigned short* cr = ql + (size_t)NQ * DK;           // raw context bf16
    unsigned short* cn = cr + (size_t)NC * DK;           // normalized context bf16
    unsigned short* wb = cn + (size_t)NC * DK;           // W bf16
    float* pm   = (float*)(wb + (size_t)DK * DK);        // chunk partial max
    float* pl   = pm + (size_t)NQ * NCHUNK;              // chunk partial sumexp
    float* pa   = pl + (size_t)NQ * NCHUNK;              // chunk partial dot
    float* yrow = pa + (size_t)NQ * NCHUNK;              // per-row result

    prep_rows<<<dim3(NQ / 8), 256, 0, stream>>>(query, qr, qn, NQ);
    prep_rows<<<dim3(NC / 8), 256, 0, stream>>>(context, cr, cn, NC);
    wconv<<<dim3(DK * DK / 256), 256, 0, stream>>>(W, wb);
    qlin_wmma<<<dim3(NQ / 128), 256, 0, stream>>>(qr, wb, ql);
    attn_flash<<<dim3(NQ / 128, NCHUNK), 256, 0, stream>>>(qn, ql, cn, cr, pm, pl, pa);
    merge_chunks<<<dim3(NQ / 256), 256, 0, stream>>>(pm, pl, pa, yrow);
    final_reduce<<<1, 256, 0, stream>>>(yrow, out);
}
